// KMax_43894565765585
// MI455X (gfx1250) — matrix operand order repaired and checked by reading, not compiled, not run on hardware
//
#include <hip/hip_runtime.h>
#include <hip/hip_bf16.h>

// Problem constants from the reference
#define BH_   32          // B*H = 2*16
#define S_    2048
#define D_    64
#define OUTW_ (S_ + 1)    // S + N_SINK

typedef __attribute__((ext_vector_type(16))) __bf16 v16bf;
typedef __attribute__((ext_vector_type(8)))  float  v8f;

// One workgroup handles a 16-row query tile for one (b,h).
// Score pass: 8 waves split the LOWER-TRIANGLE key tiles (kt <= qt) via
// v_wmma_f32_16x16x32_bf16; scores land in a 128KB LDS tile (CDNA5: 320KB/WGP).
// Softmax pass: masked-above-diagonal entries are exactly 0, so they are folded
// in analytically (count * exp(-mx)) and their outputs are a constant fill --
// no LDS reads, no v_exp for the masked half.
__launch_bounds__(256, 1)
__global__ void sink_softmax_wmma(const float* __restrict__ q,
                                  const float* __restrict__ k,
                                  const float* __restrict__ sinks,
                                  float* __restrict__ out) {
    extern __shared__ float sc[];  // 16 * 2048 floats = 128 KB

    const int tid   = threadIdx.x;
    const int lane  = tid & 31;
    // wave index is wave-invariant: pin it to an SGPR so the score loop is a
    // scalar loop (no EXEC masking around the WMMAs -> EXEC stays all-1s).
    const int wave  = __builtin_amdgcn_readfirstlane(tid >> 5);

    const int qt    = blockIdx.x & 127;   // S/16 = 128 query tiles
    const int bh    = blockIdx.x >> 7;    // 0..31
    const int qbase = qt * 16;

    const int row16 = lane & 15;          // M (A/C) or N (B/C) within 16x16 tile
    const int hi    = (lane >> 4) & 1;    // half-wave group

    // ---------------- Load A fragments (q tile row, f32 -> bf16) ----------------
    // A 16x32 layout (contiguous, matching B): lanes 0-15: K = 0..15;
    // lanes 16-31: K = 16..31. Element e -> K = hi*16 + e.
    const float* qrow = q + ((size_t)(bh * S_ + qbase + row16)) * D_;
    v16bf a0, a1;
    #pragma unroll
    for (int g = 0; g < 2; ++g) {
        #pragma unroll
        for (int o = 0; o < 8; ++o) {
            const int kk = hi * 16 + g * 8 + o;
            a0[g * 8 + o] = (__bf16)qrow[kk];
            a1[g * 8 + o] = (__bf16)qrow[32 + kk];
        }
    }

    // -------- Score pass: only lower-triangle tiles, wave w owns kt = w, w+8, ...
    // Scalar loop bounds -> EXEC is all-1s around the WMMAs (ISA requirement).
    for (int kt = wave; kt <= qt; kt += 8) {
        const int key = kt * 16 + row16;
        const float* krow = k + ((size_t)(bh * S_ + key)) * D_;

        // prefetch this wave's next key tile row (speculative, OOB-safe)
        __builtin_prefetch(krow + (size_t)8 * 16 * D_, 0, 1);

        // 16-bit B 32x16 operand (contiguous k order, as for A): N = lane%16, element e -> K = hi*16 + e.
        // k memory is [key][d] row-major == exactly this layout (no transpose).
        v16bf b0, b1;
        float ss = 0.f;
        #pragma unroll
        for (int e = 0; e < 16; ++e) {
            const float x0 = krow[hi * 16 + e];        // d in [16*hi, 16*hi+16)
            const float x1 = krow[32 + hi * 16 + e];   // d in [32+16*hi, ...)
            ss += x0 * x0 + x1 * x1;
            b0[e] = (__bf16)x0;
            b1[e] = (__bf16)x1;
        }
        // lanes L and L+16 each saw half of d=0..63 for the same key
        ss += __shfl_xor(ss, 16, 32);
        const float rinv = rsqrtf(ss);   // 1 / ||k||

        v8f acc = {};
        acc = __builtin_amdgcn_wmma_f32_16x16x32_bf16(
                  false, a0, false, b0, (short)0, acc, false, false);
        acc = __builtin_amdgcn_wmma_f32_16x16x32_bf16(
                  false, a1, false, b1, (short)0, acc, false, false);

        // C/D layout: VGPR r -> M = r + 8*hi, N = lane%16
        // Only the diagonal tile (kt == qt) has masked elements; store 0 there
        // so the softmax prefix scan below never reads garbage.
        #pragma unroll
        for (int r = 0; r < 8; ++r) {
            const int m  = r + hi * 8;
            const int mg = qbase + m;
            const float val = (key <= mg) ? acc[r] * rinv : 0.f;
            sc[m * S_ + key] = val;
        }
    }

    __syncthreads();

    // ---------------- Fused softmax: wave w handles rows 2w, 2w+1 ----------------
    const float* sinkp = sinks + (size_t)bh * S_;
    #pragma unroll
    for (int rr = 0; rr < 2; ++rr) {
        const int m  = wave * 2 + rr;      // scalar row index
        const int mg = qbase + m;          // last valid key for this row
        const int nmask = (S_ - 1) - mg;   // # of exactly-zero masked entries
        const float snk = sinkp[mg];
        const float* srow = sc + m * S_;

        // row max over valid prefix + analytic 0 for the masked run + sink
        float mx = snk;
        if (nmask > 0) mx = fmaxf(mx, 0.f);
        for (int j = lane; j <= mg; j += 32) mx = fmaxf(mx, srow[j]);
        #pragma unroll
        for (int dd = 16; dd > 0; dd >>= 1) mx = fmaxf(mx, __shfl_xor(mx, dd, 32));

        // row sum of exp over valid prefix
        float sum = 0.f;
        for (int j = lane; j <= mg; j += 32) sum += __expf(srow[j] - mx);
        #pragma unroll
        for (int dd = 16; dd > 0; dd >>= 1) sum += __shfl_xor(sum, dd, 32);
        // masked zeros contribute nmask * exp(0 - mx); sink contributes once
        sum += (float)nmask * __expf(-mx) + __expf(snk - mx);
        const float rtot = 1.f / sum;

        float* orow = out + (size_t)(bh * S_ + mg) * OUTW_;

        // valid prefix: exp-normalized scores (non-temporal 537MB stream)
        for (int j = lane; j <= mg; j += 32)
            __builtin_nontemporal_store(__expf(srow[j] - mx) * rtot, orow + j);

        // masked tail: one constant, pure store loop (no LDS read, no exp)
        const float zval = __expf(-mx) * rtot;
        const int start = (mg + 1) + ((lane - (mg + 1)) & 31); // first j>mg, j%32==lane
        for (int j = start; j < S_; j += 32)
            __builtin_nontemporal_store(zval, orow + j);

        if (lane == 0)
            __builtin_nontemporal_store(__expf(snk - mx) * rtot, orow + S_);
    }
}

extern "C" void kernel_launch(void* const* d_in, const int* in_sizes, int n_in,
                              void* d_out, int out_size, void* d_ws, size_t ws_size,
                              hipStream_t stream) {
    const float* q     = (const float*)d_in[0];
    const float* k     = (const float*)d_in[1];
    const float* sinks = (const float*)d_in[2];
    float* out = (float*)d_out;

    const dim3 grid(BH_ * (S_ / 16));   // 4096 workgroups
    const dim3 block(256);              // 8 waves (wave32)
    const size_t lds_bytes = (size_t)16 * S_ * sizeof(float);  // 128 KB

    hipLaunchKernelGGL(sink_softmax_wmma, grid, block, lds_bytes, stream,
                       q, k, sinks, out);
}
